// Attention_79671643340898
// MI455X (gfx1250) — compile-verified, matrix operand order checked by reading, not run on hardware
//
#include <hip/hip_runtime.h>
#include <hip/hip_bf16.h>
#include <stdint.h>
#include <stddef.h>

// ---------------- problem constants ----------------
#define B_    32
#define T_    784
#define C_    384
#define NH_   6
#define HD_   64
#define HW_   28
#define SCALE_ 0.05103103630798288f   // 384^-0.5
#define BN_EPS_ 1e-5f

typedef __attribute__((ext_vector_type(16))) __bf16        bf16x16;
typedef __attribute__((ext_vector_type(8)))  float         f32x8;
typedef __attribute__((ext_vector_type(4)))  unsigned int  u32x4;

union AFrag { bf16x16 v; u32x4 q[2]; };

// round-to-nearest-even f32 -> bf16 (bit twiddle; avoids relying on __bf16 scalar cvt)
__device__ __forceinline__ unsigned short f2bf(float f) {
  unsigned int u = __builtin_bit_cast(unsigned int, f);
  unsigned int lsb = (u >> 16) & 1u;
  u += 0x7fffu + lsb;
  return (unsigned short)(u >> 16);
}

// Load a 16x32 bf16 tile (row-major, stride ld elements) as a WMMA A/B fragment.
// ISA 16-bit A layout: lanes 0-15 = rows, K {0..7,16..23}; lanes 16-31 same rows,
// K {8..15,24..31}.  => two contiguous 16-byte loads per lane.
__device__ __forceinline__ bf16x16 load_frag_g(const unsigned short* __restrict__ base, int ld) {
  const int lane = threadIdx.x & 31;
  const unsigned short* p = base + (size_t)(lane & 15) * ld + ((lane >> 4) << 3);
  AFrag f;
  f.q[0] = *(const u32x4*)p;          // K = kh .. kh+7
  f.q[1] = *(const u32x4*)(p + 16);   // K = kh+16 .. kh+23
  return f.v;
}

__device__ __forceinline__ f32x8 wmma_bf16(bf16x16 a, bf16x16 b, f32x8 c) {
  return __builtin_amdgcn_wmma_f32_16x16x32_bf16(false, a, false, b, (short)0, c, false, false);
}

// ---------------- kernel 1: depthwise conv 3x3 + eval BN -> bf16 ----------------
__global__ void __launch_bounds__(256) dwconv_bn_kernel(
    const float* __restrict__ x,
    const float* __restrict__ cwq, const float* __restrict__ gq, const float* __restrict__ bq,
    const float* __restrict__ mq,  const float* __restrict__ vq,
    const float* __restrict__ cwk, const float* __restrict__ gk, const float* __restrict__ bk,
    const float* __restrict__ mk,  const float* __restrict__ vk,
    const float* __restrict__ cwv, const float* __restrict__ gv, const float* __restrict__ bv,
    const float* __restrict__ mv,  const float* __restrict__ vv,
    unsigned short* __restrict__ qt, unsigned short* __restrict__ kt,
    unsigned short* __restrict__ vt)
{
  int idx = blockIdx.x * blockDim.x + threadIdx.x;
  if (idx >= B_ * T_ * C_) return;
  int c = idx % C_;
  int t = (idx / C_) % T_;
  int b = idx / (C_ * T_);
  int hh = t / HW_, ww = t % HW_;
  const float* xb = x + (size_t)b * T_ * C_;
  float sq = 0.f, sk = 0.f, sv = 0.f;
  #pragma unroll
  for (int dh = -1; dh <= 1; ++dh) {
    int hy = hh + dh;
    #pragma unroll
    for (int dw = -1; dw <= 1; ++dw) {
      int wx = ww + dw;
      float xv = 0.f;
      if (hy >= 0 && hy < HW_ && wx >= 0 && wx < HW_)
        xv = xb[(size_t)(hy * HW_ + wx) * C_ + c];
      int wi = c * 9 + (dh + 1) * 3 + (dw + 1);
      sq += xv * cwq[wi];
      sk += xv * cwk[wi];
      sv += xv * cwv[wi];
    }
  }
  float iq = gq[c] * rsqrtf(vq[c] + BN_EPS_);
  float ik = gk[c] * rsqrtf(vk[c] + BN_EPS_);
  float iv = gv[c] * rsqrtf(vv[c] + BN_EPS_);
  qt[idx] = f2bf(sq * iq + (bq[c] - mq[c] * iq));
  kt[idx] = f2bf(sk * ik + (bk[c] - mk[c] * ik));
  vt[idx] = f2bf(sv * iv + (bv[c] - mv[c] * iv));
}

// ---------------- kernel 2: weights f32 -> bf16 ----------------
__global__ void __launch_bounds__(256) cvt_w_kernel(
    const float* __restrict__ wq, const float* __restrict__ wk,
    const float* __restrict__ wv, const float* __restrict__ wp,
    unsigned short* __restrict__ oq, unsigned short* __restrict__ ok,
    unsigned short* __restrict__ ov, unsigned short* __restrict__ op)
{
  int i = blockIdx.x * blockDim.x + threadIdx.x;
  if (i >= C_ * C_) return;
  oq[i] = f2bf(wq[i]);
  ok[i] = f2bf(wk[i]);
  ov[i] = f2bf(wv[i]);
  op[i] = f2bf(wp[i]);
}

// ---------------- kernel 3: QKV projection GEMM (D = A @ W^T) ----------------
// one wave computes a 16x64 output tile; mode 2 (V) stores transposed [b][h][d][t]
__global__ void __launch_bounds__(256) qkv_gemm_kernel(
    const unsigned short* __restrict__ qt, const unsigned short* __restrict__ kt,
    const unsigned short* __restrict__ vt,
    const unsigned short* __restrict__ wq, const unsigned short* __restrict__ wk,
    const unsigned short* __restrict__ wv,
    unsigned short* __restrict__ qb, unsigned short* __restrict__ kb,
    unsigned short* __restrict__ vT)
{
  const int mode = blockIdx.z;
  const unsigned short* A = (mode == 0) ? qt : (mode == 1) ? kt : vt;
  const unsigned short* W = (mode == 0) ? wq : (mode == 1) ? wk : wv;
  const int wid  = threadIdx.x >> 5;
  const int lane = threadIdx.x & 31;
  const int wt = blockIdx.x * 8 + wid;
  const int TILES_N = C_ / 64;            // 6
  const int tn = wt % TILES_N;
  const int tm = wt / TILES_N;            // < 1568
  const int m0 = tm * 16, n0 = tn * 64;

  f32x8 acc[4] = {{}, {}, {}, {}};
  for (int k0 = 0; k0 < C_; k0 += 32) {
    bf16x16 a = load_frag_g(A + (size_t)m0 * C_ + k0, C_);
    #pragma unroll
    for (int j = 0; j < 4; ++j) {
      bf16x16 bfr = load_frag_g(W + (size_t)(n0 + j * 16) * C_ + k0, C_);
      acc[j] = wmma_bf16(a, bfr, acc[j]);
    }
  }

  const int ncol = lane & 15;
  const int rb   = (lane >> 4) * 8;       // C-frag rows: lanes<16 -> 0..7, else 8..15
  if (mode < 2) {
    unsigned short* O = (mode == 0) ? qb : kb;
    #pragma unroll
    for (int j = 0; j < 4; ++j)
      #pragma unroll
      for (int r = 0; r < 8; ++r) {
        int m = m0 + rb + r, n = n0 + j * 16 + ncol;
        O[(size_t)m * C_ + n] = f2bf(acc[j][r]);
      }
  } else {
    #pragma unroll
    for (int j = 0; j < 4; ++j)
      #pragma unroll
      for (int r = 0; r < 8; ++r) {
        int m = m0 + rb + r, n = n0 + j * 16 + ncol;
        int bb = m / T_, tt = m % T_;
        int h = n / HD_, d = n % HD_;
        vT[((((size_t)bb * NH_ + h) * HD_) + d) * T_ + tt] = f2bf(acc[j][r]);
      }
  }
}

// ---------------- kernel 4: flash attention, one wave per 16-query tile ----------------
__global__ void __launch_bounds__(256) attn_kernel(
    const unsigned short* __restrict__ qb, const unsigned short* __restrict__ kb,
    const unsigned short* __restrict__ vT, unsigned short* __restrict__ ob)
{
  __shared__ unsigned short pbuf[8 * 16 * 32];   // per-wave P staging (1 KB each)
  const int wid  = threadIdx.x >> 5;
  const int lane = threadIdx.x & 31;
  const int wt = blockIdx.x * 8 + wid;
  const int QT = T_ / 16;                 // 49
  const int qtile = wt % QT;
  const int bh = wt / QT;
  const int b = bh / NH_, h = bh % NH_;
  const int m0 = qtile * 16;

  const unsigned short* Qb = qb + ((size_t)b * T_ + m0) * C_ + h * HD_;
  bf16x16 aq0 = load_frag_g(Qb, C_);        // d = 0..31
  bf16x16 aq1 = load_frag_g(Qb + 32, C_);   // d = 32..63

  f32x8 o0 = {}, o1 = {}, o2 = {}, o3 = {};
  float mrow[8], lrow[8];
  #pragma unroll
  for (int r = 0; r < 8; ++r) { mrow[r] = -1e30f; lrow[r] = 0.f; }

  const int ncol = lane & 15;
  const int rb   = (lane >> 4) * 8;
  unsigned short* pb = pbuf + wid * (16 * 32);
  const unsigned short* Vb = vT + (((size_t)b * NH_ + h) * HD_) * T_;

  #pragma unroll 1
  for (int t0 = 0; t0 < T_; t0 += 32) {
    // S = Q @ K^T : two C-frags (keys t0..+15, t0+16..+31)
    f32x8 s0 = {}, s1 = {};
    const unsigned short* Kb = kb + ((size_t)b * T_ + t0) * C_ + h * HD_;
    {
      bf16x16 bk0 = load_frag_g(Kb, C_);
      bf16x16 bk1 = load_frag_g(Kb + (size_t)16 * C_, C_);
      s0 = wmma_bf16(aq0, bk0, s0);
      s1 = wmma_bf16(aq0, bk1, s1);
      bk0 = load_frag_g(Kb + 32, C_);
      bk1 = load_frag_g(Kb + (size_t)16 * C_ + 32, C_);
      s0 = wmma_bf16(aq1, bk0, s0);
      s1 = wmma_bf16(aq1, bk1, s1);
    }

    // online softmax across key tile; xor masks <16 keep the two row-halves separate
    const bool v0 = (t0 + ncol)      < T_;
    const bool v1 = (t0 + 16 + ncol) < T_;
    #pragma unroll
    for (int r = 0; r < 8; ++r) {
      float a = v0 ? s0[r] * SCALE_ : -1e30f;
      float c = v1 ? s1[r] * SCALE_ : -1e30f;
      float mx = fmaxf(a, c);
      mx = fmaxf(mx, __shfl_xor(mx, 1, 32));
      mx = fmaxf(mx, __shfl_xor(mx, 2, 32));
      mx = fmaxf(mx, __shfl_xor(mx, 4, 32));
      mx = fmaxf(mx, __shfl_xor(mx, 8, 32));
      float nm   = fmaxf(mrow[r], mx);
      float corr = __expf(mrow[r] - nm);
      float p0 = __expf(a - nm);
      float p1 = __expf(c - nm);
      float rs = p0 + p1;
      rs += __shfl_xor(rs, 1, 32);
      rs += __shfl_xor(rs, 2, 32);
      rs += __shfl_xor(rs, 4, 32);
      rs += __shfl_xor(rs, 8, 32);
      lrow[r] = lrow[r] * corr + rs;
      mrow[r] = nm;
      o0[r] *= corr; o1[r] *= corr; o2[r] *= corr; o3[r] *= corr;
      // stage P (bf16) C-layout -> LDS row-major 16x32
      pb[(rb + r) * 32 + ncol]      = f2bf(p0);
      pb[(rb + r) * 32 + 16 + ncol] = f2bf(p1);
    }
    __builtin_amdgcn_wave_barrier();
    // reload P as an A-fragment (same-wave LDS ops are in-order)
    AFrag pf;
    {
      const unsigned short* pp = pb + (size_t)(lane & 15) * 32 + ((lane >> 4) << 3);
      pf.q[0] = *(const u32x4*)pp;
      pf.q[1] = *(const u32x4*)(pp + 16);
    }
    __builtin_amdgcn_wave_barrier();
    // O += P @ V ; V^T rows (fixed d) are contiguous in t
    o0 = wmma_bf16(pf.v, load_frag_g(Vb + (size_t)0 * 16 * T_ + t0, T_), o0);
    o1 = wmma_bf16(pf.v, load_frag_g(Vb + (size_t)1 * 16 * T_ + t0, T_), o1);
    o2 = wmma_bf16(pf.v, load_frag_g(Vb + (size_t)2 * 16 * T_ + t0, T_), o2);
    o3 = wmma_bf16(pf.v, load_frag_g(Vb + (size_t)3 * 16 * T_ + t0, T_), o3);
  }

  unsigned short* Ob = ob + ((size_t)b * T_ + m0) * C_ + h * HD_;
  #pragma unroll
  for (int r = 0; r < 8; ++r) {
    float inv = 1.f / lrow[r];
    Ob[(size_t)(rb + r) * C_ +  0 + ncol] = f2bf(o0[r] * inv);
    Ob[(size_t)(rb + r) * C_ + 16 + ncol] = f2bf(o1[r] * inv);
    Ob[(size_t)(rb + r) * C_ + 32 + ncol] = f2bf(o2[r] * inv);
    Ob[(size_t)(rb + r) * C_ + 48 + ncol] = f2bf(o3[r] * inv);
  }
}

// ---------------- kernel 5: output projection (+bias), f32 out ----------------
__global__ void __launch_bounds__(256) proj_gemm_kernel(
    const unsigned short* __restrict__ A, const unsigned short* __restrict__ W,
    const float* __restrict__ bias, float* __restrict__ out)
{
  const int wid  = threadIdx.x >> 5;
  const int lane = threadIdx.x & 31;
  const int wt = blockIdx.x * 8 + wid;
  const int tn = wt % (C_ / 64);
  const int tm = wt / (C_ / 64);
  const int m0 = tm * 16, n0 = tn * 64;

  f32x8 acc[4] = {{}, {}, {}, {}};
  for (int k0 = 0; k0 < C_; k0 += 32) {
    bf16x16 a = load_frag_g(A + (size_t)m0 * C_ + k0, C_);
    #pragma unroll
    for (int j = 0; j < 4; ++j) {
      bf16x16 bfr = load_frag_g(W + (size_t)(n0 + j * 16) * C_ + k0, C_);
      acc[j] = wmma_bf16(a, bfr, acc[j]);
    }
  }
  const int ncol = lane & 15;
  const int rb   = (lane >> 4) * 8;
  #pragma unroll
  for (int j = 0; j < 4; ++j) {
    int n = n0 + j * 16 + ncol;
    float bv = bias[n];
    #pragma unroll
    for (int r = 0; r < 8; ++r)
      out[(size_t)(m0 + rb + r) * C_ + n] = acc[j][r] + bv;
  }
}

// ---------------- host launcher ----------------
extern "C" void kernel_launch(void* const* d_in, const int* in_sizes, int n_in,
                              void* d_out, int out_size, void* d_ws, size_t ws_size,
                              hipStream_t stream) {
  (void)in_sizes; (void)n_in; (void)out_size; (void)ws_size;

  const float* x   = (const float*)d_in[0];
  // d_in[1]=h, d_in[2]=w (28x28 hardcoded)
  const float* cwq = (const float*)d_in[3];
  const float* gq  = (const float*)d_in[4];
  const float* bq  = (const float*)d_in[5];
  const float* mq  = (const float*)d_in[6];
  const float* vq  = (const float*)d_in[7];
  const float* cwk = (const float*)d_in[8];
  const float* gk  = (const float*)d_in[9];
  const float* bk  = (const float*)d_in[10];
  const float* mk  = (const float*)d_in[11];
  const float* vk  = (const float*)d_in[12];
  const float* cwv = (const float*)d_in[13];
  const float* gv  = (const float*)d_in[14];
  const float* bv  = (const float*)d_in[15];
  const float* mv  = (const float*)d_in[16];
  const float* vv  = (const float*)d_in[17];
  const float* wqf = (const float*)d_in[18];
  const float* wkf = (const float*)d_in[19];
  const float* wvf = (const float*)d_in[20];
  const float* wpf = (const float*)d_in[21];
  const float* bp  = (const float*)d_in[22];

  char* ws = (char*)d_ws;
  const size_t SZ  = (size_t)B_ * T_ * C_ * 2;   // 19,267,584 B per bf16 activation
  const size_t SLK = 32768;                      // slack for padded key-tile reads

  unsigned short* qt   = (unsigned short*)ws; ws += SZ;
  unsigned short* kt   = (unsigned short*)ws; ws += SZ;
  unsigned short* vt   = (unsigned short*)ws; ws += SZ;
  unsigned short* qbuf = (unsigned short*)ws; ws += SZ + SLK;
  unsigned short* kbuf = (unsigned short*)ws; ws += SZ + SLK;
  unsigned short* vTb  = (unsigned short*)ws; ws += SZ + SLK;
  unsigned short* obuf = (unsigned short*)ws; ws += SZ + SLK;
  unsigned short* wbq  = (unsigned short*)ws; ws += (size_t)C_ * C_ * 2;
  unsigned short* wbk  = (unsigned short*)ws; ws += (size_t)C_ * C_ * 2;
  unsigned short* wbv  = (unsigned short*)ws; ws += (size_t)C_ * C_ * 2;
  unsigned short* wbp  = (unsigned short*)ws; ws += (size_t)C_ * C_ * 2;

  const int n_elem = B_ * T_ * C_;
  dwconv_bn_kernel<<<(n_elem + 255) / 256, 256, 0, stream>>>(
      x, cwq, gq, bq, mq, vq, cwk, gk, bk, mk, vk, cwv, gv, bv, mv, vv, qt, kt, vt);

  cvt_w_kernel<<<(C_ * C_ + 255) / 256, 256, 0, stream>>>(
      wqf, wkf, wvf, wpf, wbq, wbk, wbv, wbp);

  // 1568 m-tiles * 6 n-tiles = 9408 wave-tiles / 8 waves per block
  const int gemm_blocks = (B_ * T_ / 16) * (C_ / 64) / 8;   // 1176
  qkv_gemm_kernel<<<dim3(gemm_blocks, 1, 3), 256, 0, stream>>>(
      qt, kt, vt, wbq, wbk, wbv, qbuf, kbuf, vTb);

  // 32*6*49 = 9408 query tiles / 8 waves per block
  const int attn_blocks = B_ * NH_ * (T_ / 16) / 8;          // 1176
  attn_kernel<<<attn_blocks, 256, 0, stream>>>(qbuf, kbuf, vTb, obuf);

  proj_gemm_kernel<<<gemm_blocks, 256, 0, stream>>>(obuf, wbp, bp, (float*)d_out);
}
